// MultiHeadAttention_73650099192316
// MI455X (gfx1250) — compile-verified, matrix operand order checked by reading, not run on hardware
//
#include <hip/hip_runtime.h>
#include <hip/hip_bf16.h>

// ---------------------------------------------------------------------------
// Windowed multi-head attention, MI455X (gfx1250), bf16 WMMA pipeline with
// CDNA5 async global->LDS staging (ASYNCcnt) and double-buffered GEMM tiles.
// Workspace (~608 MB): x_bf16, Wqkv^T_bf16, Wproj^T_bf16, qbuf, kbuf,
// vbuf(T), attout.
// ---------------------------------------------------------------------------

typedef __bf16 bf16;
typedef bf16  bf16x16 __attribute__((ext_vector_type(16)));
typedef float floatx8 __attribute__((ext_vector_type(8)));

#define L_WIN  64
#define NHEAD  8
#define DHEAD  128
#define CSIZE  512
#define TOKENS 65536
#define NWIN   (TOKENS / L_WIN)      // 1024
#define NQKV   (3 * NHEAD * DHEAD)   // 3072
#define HD     (NHEAD * DHEAD)       // 1024

static __device__ __forceinline__ bf16 f2bf(float f) { return (bf16)f; }

// Async copy of 16 contiguous bytes global -> LDS (per active lane).
// GV mode: vaddr = 64-bit VGPR pair, saddr = off.  Tracked by ASYNCcnt.
static __device__ __forceinline__ void async_copy16(void* ldsDst,
                                                    const void* gSrc) {
  unsigned lds = (unsigned)(size_t)ldsDst;  // generic->LDS: addr[31:0]
  asm volatile("global_load_async_to_lds_b128 %0, %1, off"
               :: "v"(lds), "v"(gSrc)
               : "memory");
}
static __device__ __forceinline__ void wait_async_le4() {
  asm volatile("s_wait_asynccnt 0x4" ::: "memory");
}
static __device__ __forceinline__ void wait_async_0() {
  asm volatile("s_wait_asynccnt 0x0" ::: "memory");
}

// A-fragment (and B-fragment via B^T) loader for v_wmma_f32_16x16x32_bf16.
// Row-major bf16 tile in LDS, leading dim `ld`.  Per ISA 7.12.2:
// lane -> M = lane&15; element e -> K = (e>>3)*16 + ((lane>>4)<<3) + (e&7).
static __device__ __forceinline__ bf16x16 load_frag(const bf16* p, int ld, int lane) {
  const int m  = lane & 15;
  const int kh = (lane >> 4) << 3;
  const bf16* row = p + m * ld;
  bf16x16 f;
#pragma unroll
  for (int e = 0; e < 16; e += 2) {
    const int k = ((e >> 3) << 4) + kh + (e & 7);
    f[e]     = row[k];
    f[e + 1] = row[k + 1];
  }
  return f;
}

static __device__ __forceinline__ floatx8 wmma_bf16(bf16x16 a, bf16x16 b, floatx8 c) {
  return __builtin_amdgcn_wmma_f32_16x16x32_bf16(false, a, false, b, (short)0, c,
                                                 false, false);
}

// ---------------------------------------------------------------------------
// Kernel 0a: fp32 -> bf16 convert (x)
// ---------------------------------------------------------------------------
__global__ void cvt_f32_bf16(const float* __restrict__ in, bf16* __restrict__ out,
                             int n) {
  int i = blockIdx.x * blockDim.x + threadIdx.x;
  if (i < n) out[i] = f2bf(in[i]);
}

// Kernel 0b: fp32 (K,N) -> bf16 transposed (N,K), for weights.
__global__ void cvt_f32_bf16_T(const float* __restrict__ in, bf16* __restrict__ out,
                               int K, int N) {
  int i = blockIdx.x * blockDim.x + threadIdx.x;
  if (i < K * N) {
    const int k = i / N, n = i - k * N;
    out[n * K + k] = f2bf(in[i]);
  }
}

// ---------------------------------------------------------------------------
// Kernel 1: qkv = x @ Wqkv + bqkv, fused de-interleave epilogue.
// Double-buffered async LDS staging; B read from Wqkv^T (3072, 512).
// Grid (512, 24), 256 threads = 8 waves (4x2), WG tile 128x128, K step 32.
// Per stage, per thread: 2 A-chunks + 2 B-chunks of 16B -> wait_asynccnt 4.
// ---------------------------------------------------------------------------
__global__ __launch_bounds__(256) void gemm_qkv(
    const bf16* __restrict__ xbf,   // (65536, 512)
    const bf16* __restrict__ wT,    // (3072, 512)  = Wqkv^T
    const float* __restrict__ bqkv, // (3072)
    bf16* __restrict__ qbuf, bf16* __restrict__ kbuf, bf16* __restrict__ vbuf) {
  __shared__ bf16 ldsA[2][128 * 32];   // (m, k) row-major
  __shared__ bf16 ldsB[2][128 * 32];   // B^T: (n, k) row-major

  const int tid  = threadIdx.x;
  const int lane = tid & 31;
  const int wave = tid >> 5;
  const int wm   = wave >> 1;       // 0..3
  const int wn   = wave & 1;        // 0..1
  const int row0 = blockIdx.x * 128;
  const int col0 = blockIdx.y * 128;

  floatx8 acc[2][4];
#pragma unroll
  for (int a = 0; a < 2; a++)
#pragma unroll
    for (int b = 0; b < 4; b++)
#pragma unroll
      for (int r = 0; r < 8; r++) acc[a][b][r] = 0.0f;

  auto issue_stage = [&](int buf, int kb) {
#pragma unroll
    for (int j = 0; j < 2; j++) {
      const int c = tid + 256 * j;     // chunk id 0..511 (16B chunks)
      const int r = c >> 2;            // tile row 0..127
      const int q = (c & 3) * 8;       // bf16 offset within 64B row
      async_copy16(&ldsA[buf][r * 32 + q],
                   xbf + (size_t)(row0 + r) * CSIZE + kb + q);
      async_copy16(&ldsB[buf][r * 32 + q],
                   wT + (size_t)(col0 + r) * CSIZE + kb + q);
    }
  };

  issue_stage(0, 0);
  for (int it = 0; it < CSIZE / 32; it++) {
    const int buf = it & 1;
    if (it + 1 < CSIZE / 32) {
      issue_stage(buf ^ 1, (it + 1) * 32);
      wait_async_le4();                // current stage's 4 copies complete
    } else {
      wait_async_0();
    }
    __syncthreads();

    bf16x16 af[2], bfr[4];
#pragma unroll
    for (int a = 0; a < 2; a++)
      af[a] = load_frag(&ldsA[buf][(wm * 32 + a * 16) * 32], 32, lane);
#pragma unroll
    for (int b = 0; b < 4; b++)
      bfr[b] = load_frag(&ldsB[buf][(wn * 64 + b * 16) * 32], 32, lane);
#pragma unroll
    for (int a = 0; a < 2; a++)
#pragma unroll
      for (int b = 0; b < 4; b++)
        acc[a][b] = wmma_bf16(af[a], bfr[b], acc[a][b]);
    __syncthreads();                   // done reading buf before it refills
  }

  // Epilogue: bias + scatter into q/k/v layouts.  c = h*384 + d*3 + s.
#pragma unroll
  for (int a = 0; a < 2; a++)
#pragma unroll
    for (int b = 0; b < 4; b++)
#pragma unroll
      for (int r = 0; r < 8; r++) {
        const int mloc = wm * 32 + a * 16 + r + ((lane >> 4) << 3);
        const int nloc = wn * 64 + b * 16 + (lane & 15);
        const int t = row0 + mloc;
        const int c = col0 + nloc;
        const float v = acc[a][b][r] + bqkv[c];
        const int h  = c / 384;
        const int rr = c - h * 384;
        const int d  = rr / 3;
        const int s  = rr - d * 3;
        const int win = t >> 6, l = t & 63;
        const int base = win * NHEAD + h;
        if (s == 0)       qbuf[(base * 64 + l) * 128 + d] = f2bf(v);
        else if (s == 1)  kbuf[(base * 64 + l) * 128 + d] = f2bf(v);
        else              vbuf[(base * 128 + d) * 64 + l] = f2bf(v);  // V^T
      }
}

// ---------------------------------------------------------------------------
// Kernel 2: fused attention per (window, head).  128 threads = 4 waves.
// Q/K/V^T tiles stream in via async-to-LDS.  S = Q K^T; softmax; /sqrt(D);
// O = P V.  LDS reuse: S(f32) overlays Q, P(bf16) overlays K.
// ---------------------------------------------------------------------------
__global__ __launch_bounds__(128) void attn_win(
    const bf16* __restrict__ qbuf, const bf16* __restrict__ kbuf,
    const bf16* __restrict__ vbuf, bf16* __restrict__ attout) {
  __shared__ bf16 ldsQ[64 * 128];   // 16KB, later S (f32 64x64)
  __shared__ bf16 ldsK[64 * 128];   // 16KB, later P (bf16 64x64) in first half
  __shared__ bf16 ldsV[128 * 64];   // 16KB, V^T (d, key)

  float* ldsS = (float*)ldsQ;
  bf16*  ldsP = ldsK;

  const int tid  = threadIdx.x;
  const int lane = tid & 31;
  const int wave = tid >> 5;        // 0..3, owns rows 16*wave..16*wave+15
  const int wh   = blockIdx.x;      // win*8 + h
  const size_t tileoff = (size_t)wh * (64 * 128);

  // Stage 3 x 16KB tiles: 1024 chunks of 16B each, 128 threads -> 8 per tile.
#pragma unroll
  for (int j = 0; j < 8; j++) {
    const int c = (tid + 128 * j) * 8;  // bf16 element offset, 16B aligned
    async_copy16(&ldsQ[c], qbuf + tileoff + c);
    async_copy16(&ldsK[c], kbuf + tileoff + c);
    async_copy16(&ldsV[c], vbuf + tileoff + c);
  }
  wait_async_0();
  __syncthreads();

  // --- S = Q K^T : wave computes 16x64 strip, K-dim 128 in 4 steps ---
  floatx8 sacc[4];
#pragma unroll
  for (int b = 0; b < 4; b++)
#pragma unroll
    for (int r = 0; r < 8; r++) sacc[b][r] = 0.0f;

#pragma unroll
  for (int kk = 0; kk < 4; kk++) {
    const int k0 = kk * 32;
    bf16x16 a = load_frag(ldsQ + (wave * 16) * 128 + k0, 128, lane);
#pragma unroll
    for (int b = 0; b < 4; b++) {
      bf16x16 bb = load_frag(ldsK + (b * 16) * 128 + k0, 128, lane); // B^T = K
      sacc[b] = wmma_bf16(a, bb, sacc[b]);
    }
  }
  // Store S rows (own rows only -> safely overlays own Q rows)
#pragma unroll
  for (int b = 0; b < 4; b++)
#pragma unroll
    for (int r = 0; r < 8; r++) {
      const int m = wave * 16 + r + ((lane >> 4) << 3);
      ldsS[m * 64 + b * 16 + (lane & 15)] = sacc[b][r];
    }
  __syncthreads();

  // --- softmax per row, then divide by sqrt(D) (after softmax, faithful) ---
  if (tid < 64) {
    const float* srow = ldsS + tid * 64;
    float mx = -3.402823466e38f;
    for (int c = 0; c < 64; c++) mx = fmaxf(mx, srow[c]);
    float sum = 0.0f;
    for (int c = 0; c < 64; c++) sum += __expf(srow[c] - mx);
    const float inv = 1.0f / (sum * sqrtf(128.0f));
    for (int c = 0; c < 64; c++)
      ldsP[tid * 64 + c] = f2bf(__expf(srow[c] - mx) * inv);
  }
  __syncthreads();

  // --- O = P @ V : wave computes 16x128 strip, K-dim 64 in 2 steps ---
  floatx8 oacc[8];
#pragma unroll
  for (int b = 0; b < 8; b++)
#pragma unroll
    for (int r = 0; r < 8; r++) oacc[b][r] = 0.0f;

#pragma unroll
  for (int kk = 0; kk < 2; kk++) {
    const int k0 = kk * 32;
    bf16x16 a = load_frag(ldsP + (wave * 16) * 64 + k0, 64, lane);
#pragma unroll
    for (int b = 0; b < 8; b++) {
      bf16x16 bb = load_frag(ldsV + (b * 16) * 64 + k0, 64, lane); // B^T = V^T
      oacc[b] = wmma_bf16(a, bb, oacc[b]);
    }
  }

  // Epilogue: attout (T, H*D) bf16, att position (win, l, h, d)
  const int win = wh >> 3, h = wh & 7;
#pragma unroll
  for (int b = 0; b < 8; b++)
#pragma unroll
    for (int r = 0; r < 8; r++) {
      const int m = wave * 16 + r + ((lane >> 4) << 3);
      const int t = win * 64 + m;
      const int col = h * 128 + b * 16 + (lane & 15);
      attout[(size_t)t * HD + col] = f2bf(oacc[b][r]);
    }
}

// ---------------------------------------------------------------------------
// Kernel 3: out = attout @ Wproj + bproj  (65536x1024 @ 1024x128, f32 out)
// Double-buffered async staging; B from Wproj^T (128, 1024).
// Grid 512, 256 threads = 8 waves; wave owns 16 rows x 128 cols.
// ---------------------------------------------------------------------------
__global__ __launch_bounds__(256) void gemm_proj(
    const bf16* __restrict__ attout,  // (65536, 1024)
    const bf16* __restrict__ wpT,     // (128, 1024) = Wproj^T
    const float* __restrict__ bproj,  // (128)
    float* __restrict__ out) {        // (65536, 128)
  __shared__ bf16 ldsA[2][128 * 32];
  __shared__ bf16 ldsB[2][128 * 32];  // B^T: (n, k)

  const int tid  = threadIdx.x;
  const int lane = tid & 31;
  const int wave = tid >> 5;          // 0..7
  const int row0 = blockIdx.x * 128;

  floatx8 acc[8];
#pragma unroll
  for (int b = 0; b < 8; b++)
#pragma unroll
    for (int r = 0; r < 8; r++) acc[b][r] = 0.0f;

  auto issue_stage = [&](int buf, int kb) {
#pragma unroll
    for (int j = 0; j < 2; j++) {
      const int c = tid + 256 * j;     // chunk id 0..511
      const int r = c >> 2;
      const int q = (c & 3) * 8;
      async_copy16(&ldsA[buf][r * 32 + q],
                   attout + (size_t)(row0 + r) * HD + kb + q);
      async_copy16(&ldsB[buf][r * 32 + q],
                   wpT + (size_t)r * HD + kb + q);
    }
  };

  issue_stage(0, 0);
  for (int it = 0; it < HD / 32; it++) {
    const int buf = it & 1;
    if (it + 1 < HD / 32) {
      issue_stage(buf ^ 1, (it + 1) * 32);
      wait_async_le4();
    } else {
      wait_async_0();
    }
    __syncthreads();

    bf16x16 a = load_frag(&ldsA[buf][(wave * 16) * 32], 32, lane);
#pragma unroll
    for (int b = 0; b < 8; b++) {
      bf16x16 bb = load_frag(&ldsB[buf][(b * 16) * 32], 32, lane);
      acc[b] = wmma_bf16(a, bb, acc[b]);
    }
    __syncthreads();
  }

#pragma unroll
  for (int b = 0; b < 8; b++)
#pragma unroll
    for (int r = 0; r < 8; r++) {
      const int m = wave * 16 + r + ((lane >> 4) << 3);
      const int n = b * 16 + (lane & 15);
      out[(row0 + m) * 128 + n] = acc[b][r] + bproj[n];
    }
}

// ---------------------------------------------------------------------------
extern "C" void kernel_launch(void* const* d_in, const int* in_sizes, int n_in,
                              void* d_out, int out_size, void* d_ws, size_t ws_size,
                              hipStream_t stream) {
  const float* x     = (const float*)d_in[0];
  const float* Wqkv  = (const float*)d_in[1];
  const float* bqkv  = (const float*)d_in[2];
  const float* Wproj = (const float*)d_in[3];
  const float* bproj = (const float*)d_in[4];
  float* out = (float*)d_out;

  char* ws = (char*)d_ws;
  size_t off = 0;
  auto carve = [&](size_t bytes) -> char* {
    char* p = ws + off;
    off += (bytes + 255) & ~(size_t)255;
    return p;
  };
  bf16* xbf    = (bf16*)carve((size_t)TOKENS * CSIZE * 2);
  bf16* wqT    = (bf16*)carve((size_t)CSIZE * NQKV * 2);   // (3072, 512)
  bf16* wpT    = (bf16*)carve((size_t)HD * DHEAD * 2);     // (128, 1024)
  bf16* qbuf   = (bf16*)carve((size_t)NWIN * NHEAD * L_WIN * DHEAD * 2);
  bf16* kbuf   = (bf16*)carve((size_t)NWIN * NHEAD * L_WIN * DHEAD * 2);
  bf16* vbuf   = (bf16*)carve((size_t)NWIN * NHEAD * L_WIN * DHEAD * 2);
  bf16* attout = (bf16*)carve((size_t)TOKENS * HD * 2);

  const int nx = TOKENS * CSIZE;
  const int nw = CSIZE * NQKV;
  const int np = HD * DHEAD;
  cvt_f32_bf16<<<(nx + 255) / 256, 256, 0, stream>>>(x, xbf, nx);
  cvt_f32_bf16_T<<<(nw + 255) / 256, 256, 0, stream>>>(Wqkv, wqT, CSIZE, NQKV);
  cvt_f32_bf16_T<<<(np + 255) / 256, 256, 0, stream>>>(Wproj, wpT, HD, DHEAD);

  gemm_qkv<<<dim3(TOKENS / 128, NQKV / 128), 256, 0, stream>>>(
      xbf, wqT, bqkv, qbuf, kbuf, vbuf);

  attn_win<<<NWIN * NHEAD, 128, 0, stream>>>(qbuf, kbuf, vbuf, attout);

  gemm_proj<<<TOKENS / 128, 256, 0, stream>>>(attout, wpT, bproj, out);
}
